// DGLJTNNDecoder_69853348102867
// MI455X (gfx1250) — matrix operand order repaired and checked by reading, not compiled, not run on hardware
//
#include <hip/hip_runtime.h>
#include <stdint.h>
#include <stddef.h>

// Problem constants (match reference)
#define B_   1024
#define L_   24
#define H_   512
#define LAT_ 128
#define V_   800
#define E_   46   // 2*(L-1)
#define T_   46

typedef __attribute__((ext_vector_type(16))) __bf16 v16bf;
typedef __attribute__((ext_vector_type(8)))  float  v8f;
typedef __attribute__((ext_vector_type(4)))  unsigned int u32x4;

union FragBF { u32x4 q[2]; v16bf v; };
union FragF  { v8f v; float f[8]; };

#if defined(__gfx1250__) && __has_builtin(__builtin_amdgcn_global_load_async_to_lds_b128)
#define HAVE_ASYNC_LDS 1
#else
#define HAVE_ASYNC_LDS 0
#endif

#if HAVE_ASYNC_LDS
typedef int v4i_vs __attribute__((vector_size(4 * sizeof(int))));
typedef __attribute__((address_space(1))) v4i_vs* async_gptr_t;   // global src
typedef __attribute__((address_space(3))) v4i_vs* async_lptr_t;   // LDS dst
#endif

__device__ __forceinline__ unsigned short f2bf(float x) {
  unsigned int u = __float_as_uint(x);
  u = (u + 0x7FFFu + ((u >> 16) & 1u)) >> 16;   // round-to-nearest-even
  return (unsigned short)u;
}
__device__ __forceinline__ float bf2f(unsigned short h) {
  return __uint_as_float(((unsigned int)h) << 16);
}
__device__ __forceinline__ float sigm(float x) { return 1.0f / (1.0f + expf(-x)); }

__device__ __forceinline__ int dst_of_step(int t) {
  return (t < L_ - 1) ? (t + 1) : (L_ - 2 - (t - (L_ - 1)));
}

// 16-byte global -> LDS stage; async (ASYNCcnt-tracked) when available.
__device__ __forceinline__ void stage16(const unsigned short* g, unsigned short* l) {
#if HAVE_ASYNC_LDS
  __builtin_amdgcn_global_load_async_to_lds_b128(
      (async_gptr_t)g, (async_lptr_t)l, 0, 0);
#else
  *(u32x4*)l = *(const u32x4*)g;
#endif
}
__device__ __forceinline__ void async_wait_all() {
#if HAVE_ASYNC_LDS
#if __has_builtin(__builtin_amdgcn_s_wait_asynccnt)
  __builtin_amdgcn_s_wait_asynccnt(0);
#else
  asm volatile("s_wait_asynccnt 0" ::: "memory");
#endif
#endif
}

// ---------------------------------------------------------------------------
// WMMA GEMM: C[M,N] = act(A[M,K](bf16 row-major) * W^T[N,K](bf16 row-major) + bias)
// M % 256 == 0, N % 32 == 0, K % 32 == 0 (all call sites satisfy this).
// 256 threads = 8 wave32; macro tile 256x32; wave tile 32x32 (2x2 WMMA 16x16).
// Double-buffered LDS, async global->LDS staging, one barrier per K-step.
// ---------------------------------------------------------------------------
__device__ __forceinline__ v16bf fragA(const unsigned short* As, int rowBase, int lane) {
  const int m  = lane & 15;
  const int kh = (lane >> 4) << 3;          // 0 or 8
  const unsigned short* p = As + (rowBase + m) * 32;
  FragBF f;
  f.q[0] = *(const u32x4*)(p + kh);         // K = kh..kh+7
  f.q[1] = *(const u32x4*)(p + 16 + kh);    // K = 16+kh..16+kh+7
  return f.v;
}
__device__ __forceinline__ v16bf fragB(const unsigned short* Bt, int nBase, int lane) {
  const int n  = lane & 15;
  const int kg = (lane >> 4) << 3;          // 0 or 8
  const unsigned short* p = Bt + (nBase + n) * 32 + kg;
  FragBF f;
  f.q[0] = *(const u32x4*)(p);
  f.q[1] = *(const u32x4*)(p + 16);
  return f.v;
}

__global__ __launch_bounds__(256) void k_gemm(
    const unsigned short* __restrict__ A, int lda,     // [M][K], lda in elements
    const unsigned short* __restrict__ BT,             // [N][K] row-major (ld = K)
    int K,
    const float* __restrict__ bias, int relu,
    float* __restrict__ Cf, unsigned short* __restrict__ Cb, int ldc)
{
  __shared__ __align__(16) unsigned short As[2][256 * 32];
  __shared__ __align__(16) unsigned short Bt[2][32 * 32];

  const int tid  = threadIdx.x;
  const int lane = tid & 31;
  const int wave = tid >> 5;                 // 0..7 -> rows wave*32..+32
  const int m0 = blockIdx.x * 256;
  const int n0 = blockIdx.y * 32;

  v8f c00 = {0,0,0,0,0,0,0,0};
  v8f c01 = c00, c10 = c00, c11 = c00;

  const unsigned short* arow = A + (size_t)(m0 + tid) * lda;   // this thread's A row
  const int bn = tid >> 2;                   // 0..63 (use 0..31)
  const int bo = (tid & 3) * 8;              // 0,8,16,24

  auto stageA = [&](int buf, int k0) {
    unsigned short* l = &As[buf][tid * 32];
    const unsigned short* g = arow + k0;
    stage16(g,      l);
    stage16(g + 8,  l + 8);
    stage16(g + 16, l + 16);
    stage16(g + 24, l + 24);
  };
  auto stageB = [&](int buf, int k0) {
    if (tid < 128)
      stage16(BT + (size_t)(n0 + bn) * K + k0 + bo, &Bt[buf][bn * 32 + bo]);
  };

  const int nk = K >> 5;
  stageA(0, 0);
  stageB(0, 0);

  for (int i = 0; i < nk; ++i) {
    async_wait_all();
    __syncthreads();                          // tile i resident; tile i-1 reads done
    if (i + 1 < nk) {                         // overlap: stage tile i+1 into other buffer
      stageA((i + 1) & 1, (i + 1) * 32);
      stageB((i + 1) & 1, (i + 1) * 32);
    }
    if (i + 2 < nk)
      __builtin_prefetch((const void*)(arow + (i + 2) * 32), 0, 3);

    const unsigned short* as = As[i & 1];
    const unsigned short* bt = Bt[i & 1];
    v16bf a0 = fragA(as, wave * 32,      lane);
    v16bf a1 = fragA(as, wave * 32 + 16, lane);
    v16bf b0 = fragB(bt, 0,  lane);
    v16bf b1 = fragB(bt, 16, lane);
    c00 = __builtin_amdgcn_wmma_f32_16x16x32_bf16(false, a0, false, b0, (short)0, c00, false, false);
    c01 = __builtin_amdgcn_wmma_f32_16x16x32_bf16(false, a0, false, b1, (short)0, c01, false, false);
    c10 = __builtin_amdgcn_wmma_f32_16x16x32_bf16(false, a1, false, b0, (short)0, c10, false, false);
    c11 = __builtin_amdgcn_wmma_f32_16x16x32_bf16(false, a1, false, b1, (short)0, c11, false, false);
  }

  v8f accs[4] = {c00, c01, c10, c11};
  #pragma unroll
  for (int idx = 0; idx < 4; ++idx) {
    const int rt = idx >> 1, ct = idx & 1;
    const int col = n0 + ct * 16 + (lane & 15);
    const float bv = bias ? bias[col] : 0.0f;
    FragF f; f.v = accs[idx];
    const int rbase = m0 + wave * 32 + rt * 16 + ((lane >> 4) << 3);
    #pragma unroll
    for (int j = 0; j < 8; ++j) {
      float v = f.f[j] + bv;
      if (relu && v < 0.0f) v = 0.0f;
      size_t o = (size_t)(rbase + j) * ldc + col;
      if (Cf) Cf[o] = v; else Cb[o] = f2bf(v);
    }
  }
}

// ---------------------------------------------------------------------------
// Elementwise / gather kernels
// ---------------------------------------------------------------------------
__global__ void k_fill_u16(unsigned short* p, int n) {
  int i = blockIdx.x * blockDim.x + threadIdx.x;
  if (i < n) p[i] = 0;
}
__global__ void k_fill_f32(float* p, int n) {
  int i = blockIdx.x * blockDim.x + threadIdx.x;
  if (i < n) p[i] = 0.0f;
}
// Transposing f32 -> bf16 weight convert: s[Ksrc][N] -> d[n][koff + k] (row len ldd)
__global__ void k_cvtT(const float* __restrict__ s, unsigned short* __restrict__ d,
                       int Ksrc, int N, int ldd, int koff) {
  int i = blockIdx.x * blockDim.x + threadIdx.x;
  if (i >= Ksrc * N) return;
  int k = i % Ksrc, n = i / Ksrc;
  d[(size_t)n * ldd + koff + k] = f2bf(s[(size_t)k * N + n]);
}
__global__ void k_embed(const int* __restrict__ wid, const float* __restrict__ emb,
                        unsigned short* __restrict__ xbf, int total) {
  int i = blockIdx.x * blockDim.x + threadIdx.x;
  if (i >= total) return;
  int h = i % H_;
  int bl = i / H_;                 // b*L + l
  xbf[i] = f2bf(emb[(size_t)wid[bl] * H_ + h]);
}
__global__ void k_concat(const unsigned short* __restrict__ left, int lstride,
                         const unsigned short* __restrict__ right, int rstride,
                         unsigned short* __restrict__ A, int total) {
  int i = blockIdx.x * blockDim.x + threadIdx.x;
  if (i >= total) return;
  int c = i & 1023;                // row length = 2H = 1024
  int b = i >> 10;
  unsigned short v;
  if (c < H_)      v = left[(size_t)b * lstride + c];
  else if (right)  v = right[(size_t)b * rstride + (c - H_)];
  else             v = 0;
  A[i] = v;
}
__global__ void k_gru1(const float* __restrict__ gz, const float* __restrict__ gh,
                       const float* __restrict__ bz, const float* __restrict__ bh,
                       unsigned short* __restrict__ m, float* __restrict__ tmpMn,
                       int eid, int pred, int total) {
  int i = blockIdx.x * blockDim.x + threadIdx.x;
  if (i >= total) return;
  int h = i & (H_ - 1);
  int b = i >> 9;
  float z  = sigm(gz[i] + bz[h]);
  float s  = (pred >= 0) ? bf2f(m[((size_t)b * E_ + pred) * H_ + h]) : 0.0f;
  float mn = (1.0f - z) * s + z * tanhf(gh[i] + bh[h]);
  m[((size_t)b * E_ + eid) * H_ + h] = f2bf(mn);
  tmpMn[i] = mn;
}
__global__ void k_gru2(const float* __restrict__ gr, const float* __restrict__ bur,
                       const float* __restrict__ tmpMn, unsigned short* __restrict__ rm,
                       int eid, int total) {
  int i = blockIdx.x * blockDim.x + threadIdx.x;
  if (i >= total) return;
  int h = i & (H_ - 1);
  int b = i >> 9;
  float r = sigm(gr[i] + bur[h]);
  rm[((size_t)b * E_ + eid) * H_ + h] = f2bf(r * tmpMn[i]);
}
__global__ void k_hs(const unsigned short* __restrict__ m, unsigned short* __restrict__ hs_t,
                     int in0, int in1, int total) {
  int i = blockIdx.x * blockDim.x + threadIdx.x;
  if (i >= total) return;
  int h = i & (H_ - 1);
  int b = i >> 9;
  float a = (in0 >= 0) ? bf2f(m[((size_t)b * E_ + in0) * H_ + h]) : 0.0f;
  float c = (in1 >= 0) ? bf2f(m[((size_t)b * E_ + in1) * H_ + h]) : 0.0f;
  hs_t[i] = f2bf(a + c);
}
__global__ void k_build_qin(const unsigned short* __restrict__ hs, const float* __restrict__ tv,
                            unsigned short* __restrict__ qin, int total) {
  int i = blockIdx.x * blockDim.x + threadIdx.x;
  if (i >= total) return;
  const int KQ = H_ + LAT_;
  int c = i % KQ;
  int r = i / KQ;
  int l = r / B_, b = r % B_;
  unsigned short v;
  if (c < H_) v = (l == 0) ? (unsigned short)0 : hs[((size_t)(l - 1) * B_ + b) * H_ + c];
  else        v = f2bf(tv[(size_t)b * LAT_ + (c - H_)]);
  qin[i] = v;
}
__global__ void k_build_pin(const unsigned short* __restrict__ xbf,
                            const unsigned short* __restrict__ hs,
                            const float* __restrict__ tv,
                            unsigned short* __restrict__ pin, int total) {
  int i = blockIdx.x * blockDim.x + threadIdx.x;
  if (i >= total) return;
  const int KP = 2 * H_ + LAT_;
  int c = i % KP;
  int r = i / KP;
  int t = r / B_, b = r % B_;
  unsigned short v;
  if (c < H_) {
    int node = (t == 0) ? 0 : dst_of_step(t - 1);
    v = xbf[((size_t)b * L_ + node) * H_ + c];
  } else if (c < 2 * H_) {
    v = (t == 0) ? (unsigned short)0 : hs[((size_t)(t - 1) * B_ + b) * H_ + (c - H_)];
  } else {
    v = f2bf(tv[(size_t)b * LAT_ + (c - 2 * H_)]);
  }
  pin[i] = v;
}
__global__ __launch_bounds__(256) void k_pfinal(const float* __restrict__ ph,
                                                const float* __restrict__ Us,
                                                const float* __restrict__ bs,
                                                float* __restrict__ accum, int rows) {
  int wave = threadIdx.x >> 5, lane = threadIdx.x & 31;
  int row = blockIdx.x * 8 + wave;
  if (row >= rows) return;
  const float* p = ph + (size_t)row * H_;
  float s = 0.0f;
  for (int c = lane; c < H_; c += 32) s += p[c] * Us[c];
  for (int m = 16; m; m >>= 1) s += __shfl_xor(s, m, 32);
  if (lane == 0) {
    float pv = s + bs[0];
    int t = row / B_;
    float pt = (t < L_ - 1) ? 1.0f : 0.0f;
    float loss = fmaxf(pv, 0.0f) - pv * pt + log1pf(expf(-fabsf(pv)));
    float acc = ((pv > 0.0f ? 1.0f : 0.0f) == pt) ? 1.0f : 0.0f;
    atomicAdd(&accum[1], loss);
    atomicAdd(&accum[3], acc);
  }
}
__global__ __launch_bounds__(256) void k_qfinal(const float* __restrict__ qlog,
                                                const int* __restrict__ wid,
                                                float* __restrict__ accum, int rows) {
  int wave = threadIdx.x >> 5, lane = threadIdx.x & 31;
  int row = blockIdx.x * 8 + wave;
  if (row >= rows) return;
  const float* q = qlog + (size_t)row * V_;
  float mv = -3.4e38f; int mi = 0;
  for (int c = lane; c < V_; c += 32) {
    float v = q[c];
    if (v > mv) { mv = v; mi = c; }
  }
  for (int m = 16; m; m >>= 1) {
    float ov = __shfl_xor(mv, m, 32);
    int   oi = __shfl_xor(mi, m, 32);
    if (ov > mv || (ov == mv && oi < mi)) { mv = ov; mi = oi; }
  }
  float se = 0.0f;
  for (int c = lane; c < V_; c += 32) se += expf(q[c] - mv);
  for (int m = 16; m; m >>= 1) se += __shfl_xor(se, m, 32);
  if (lane == 0) {
    int l = row / B_, b = row % B_;
    int tgt = wid[b * L_ + l];
    float loss = (mv + logf(se)) - q[tgt];
    atomicAdd(&accum[0], loss);
    atomicAdd(&accum[2], (mi == tgt) ? 1.0f : 0.0f);
  }
}
__global__ void k_finalize(const float* __restrict__ accum, float* __restrict__ out) {
  if (threadIdx.x == 0 && blockIdx.x == 0) {
    out[0] = accum[0] / (float)B_;                 // q_loss
    out[1] = accum[1] / (float)B_;                 // p_loss
    out[2] = accum[2] / (float)(L_ * B_);          // q_acc
    out[3] = accum[3] / (float)((T_ + 1) * B_);    // p_acc
  }
}

// ---------------------------------------------------------------------------
// Host side
// ---------------------------------------------------------------------------
static void sched_host(int t, int& eid, int& u, int& v, int& pred, int& in0, int& in1) {
  if (t < L_ - 1) { eid = t; u = t; v = t + 1; pred = t - 1; }
  else {
    int bb = (L_ - 2) - (t - (L_ - 1));
    eid = (L_ - 1) + bb; u = bb + 1; v = bb;
    pred = (bb < L_ - 2) ? (L_ - 1) + bb + 1 : -1;
  }
  in0 = (v > 0)      ? v - 1        : -1;
  in1 = (v < L_ - 1) ? (L_ - 1) + v : -1;
}

extern "C" void kernel_launch(void* const* d_in, const int* in_sizes, int n_in,
                              void* d_out, int out_size, void* d_ws, size_t ws_size,
                              hipStream_t stream) {
  (void)in_sizes; (void)n_in; (void)out_size; (void)ws_size;
  const int*   wid      = (const int*)  d_in[0];
  const float* tree_vec = (const float*)d_in[1];
  const float* emb      = (const float*)d_in[2];
  const float* Wz  = (const float*)d_in[3];
  const float* bz  = (const float*)d_in[4];
  const float* Wr  = (const float*)d_in[5];
  const float* Ur  = (const float*)d_in[6];
  const float* bur = (const float*)d_in[7];
  const float* Wh  = (const float*)d_in[8];
  const float* bh  = (const float*)d_in[9];
  const float* Ww  = (const float*)d_in[10];
  const float* bw  = (const float*)d_in[11];
  const float* Wu  = (const float*)d_in[12];
  const float* bu  = (const float*)d_in[13];
  const float* Wo  = (const float*)d_in[14];
  const float* bo  = (const float*)d_in[15];
  const float* Us  = (const float*)d_in[16];
  const float* bs  = (const float*)d_in[17];
  float* out = (float*)d_out;

  char* w = (char*)d_ws;
  size_t off = 0;
  auto alloc = [&](size_t bytes) -> void* {
    void* p = w + off;
    off = (off + bytes + 255) & ~(size_t)255;
    return p;
  };

  float*          accum = (float*)alloc(4 * sizeof(float));
  unsigned short* xbf   = (unsigned short*)alloc((size_t)B_ * L_ * H_ * 2);
  unsigned short* mB    = (unsigned short*)alloc((size_t)B_ * E_ * H_ * 2);
  unsigned short* rmB   = (unsigned short*)alloc((size_t)B_ * E_ * H_ * 2);
  unsigned short* hs    = (unsigned short*)alloc((size_t)T_ * B_ * H_ * 2);
  unsigned short* WzT   = (unsigned short*)alloc((size_t)512 * 1024 * 2);   // [N][K]
  unsigned short* WhT   = (unsigned short*)alloc((size_t)512 * 1024 * 2);
  unsigned short* WruT  = (unsigned short*)alloc((size_t)512 * 1024 * 2);   // [Wr;Ur]^T
  unsigned short* WwT   = (unsigned short*)alloc((size_t)512 * 640 * 2);
  unsigned short* WuT   = (unsigned short*)alloc((size_t)512 * 1152 * 2);
  unsigned short* WoT   = (unsigned short*)alloc((size_t)800 * 512 * 2);
  unsigned short* A1    = (unsigned short*)alloc((size_t)B_ * 1024 * 2);
  unsigned short* A2    = (unsigned short*)alloc((size_t)B_ * 1024 * 2);
  unsigned short* A3    = (unsigned short*)alloc((size_t)B_ * 1024 * 2);
  float*          gz    = (float*)alloc((size_t)B_ * H_ * 4);
  float*          gh    = (float*)alloc((size_t)B_ * H_ * 4);
  float*          gr    = (float*)alloc((size_t)B_ * H_ * 4);
  float*          tmpMn = (float*)alloc((size_t)B_ * H_ * 4);
  unsigned short* qin   = (unsigned short*)alloc((size_t)(L_ * B_) * 640 * 2);
  unsigned short* qh    = (unsigned short*)alloc((size_t)(L_ * B_) * 512 * 2);
  float*          qlog  = (float*)alloc((size_t)(L_ * B_) * 800 * 4);
  unsigned short* pin   = (unsigned short*)alloc((size_t)((T_ + 1) * B_) * 1152 * 2);
  float*          ph    = (float*)alloc((size_t)((T_ + 1) * B_) * 512 * 4);

  auto blocks = [](int n) { return dim3((unsigned)((n + 255) / 256)); };
  const dim3 TPB(256);

  auto gemm = [&](const unsigned short* A, int lda, const unsigned short* WT, int N, int M,
                  int K, const float* bias, int relu, float* Cf, unsigned short* Cb) {
    dim3 g((unsigned)(M / 256), (unsigned)(N / 32));
    k_gemm<<<g, dim3(256), 0, stream>>>(A, lda, WT, K, bias, relu, Cf, Cb, N);
  };

  // ---- init state / convert (and transpose) weights ----
  k_fill_f32<<<blocks(4), TPB, 0, stream>>>(accum, 4);
  {
    int n = B_ * E_ * H_;
    k_fill_u16<<<blocks(n), TPB, 0, stream>>>(mB, n);
    k_fill_u16<<<blocks(n), TPB, 0, stream>>>(rmB, n);
  }
  k_cvtT<<<blocks(1024 * 512), TPB, 0, stream>>>(Wz, WzT, 1024, 512, 1024, 0);
  k_cvtT<<<blocks(1024 * 512), TPB, 0, stream>>>(Wh, WhT, 1024, 512, 1024, 0);
  k_cvtT<<<blocks(512 * 512),  TPB, 0, stream>>>(Wr, WruT, 512, 512, 1024, 0);
  k_cvtT<<<blocks(512 * 512),  TPB, 0, stream>>>(Ur, WruT, 512, 512, 1024, 512);
  k_cvtT<<<blocks(640 * 512),  TPB, 0, stream>>>(Ww, WwT, 640, 512, 640, 0);
  k_cvtT<<<blocks(1152 * 512), TPB, 0, stream>>>(Wu, WuT, 1152, 512, 1152, 0);
  k_cvtT<<<blocks(512 * 800),  TPB, 0, stream>>>(Wo, WoT, 512, 800, 512, 0);
  k_embed<<<blocks(B_ * L_ * H_), TPB, 0, stream>>>(wid, emb, xbf, B_ * L_ * H_);

  // ---- sequential DFS scan ----
  const int tot2 = B_ * 1024;
  const int tot1 = B_ * H_;
  for (int t = 0; t < T_; ++t) {
    int eid, u, v, pred, in0, in1;
    sched_host(t, eid, u, v, pred, in0, in1);

    const unsigned short* mpred  = (pred >= 0) ? (mB  + (size_t)pred * H_) : nullptr;
    const unsigned short* rmpred = (pred >= 0) ? (rmB + (size_t)pred * H_) : nullptr;

    // z gate: [x_u | m_pred] @ Wz
    k_concat<<<blocks(tot2), TPB, 0, stream>>>(xbf + (size_t)u * H_, L_ * H_, mpred, E_ * H_, A1, tot2);
    gemm(A1, 1024, WzT, 512, B_, 1024, nullptr, 0, gz, nullptr);
    // candidate: [x_u | rm_pred] @ Wh
    k_concat<<<blocks(tot2), TPB, 0, stream>>>(xbf + (size_t)u * H_, L_ * H_, rmpred, E_ * H_, A2, tot2);
    gemm(A2, 1024, WhT, 512, B_, 1024, nullptr, 0, gh, nullptr);
    // GRU combine -> m[:, eid]
    k_gru1<<<blocks(tot1), TPB, 0, stream>>>(gz, gh, bz, bh, mB, tmpMn, eid, pred, tot1);
    // readout gate: [x_v | m_new] @ [Wr; Ur]
    k_concat<<<blocks(tot2), TPB, 0, stream>>>(xbf + (size_t)v * H_, L_ * H_,
                                               mB + (size_t)eid * H_, E_ * H_, A3, tot2);
    gemm(A3, 1024, WruT, 512, B_, 1024, nullptr, 0, gr, nullptr);
    k_gru2<<<blocks(tot1), TPB, 0, stream>>>(gr, bur, tmpMn, rmB, eid, tot1);
    // node pull h_v = m[in0] + m[in1]
    k_hs<<<blocks(tot1), TPB, 0, stream>>>(mB, hs + (size_t)t * B_ * H_, in0, in1, tot1);
  }

  // ---- q head: relu(q_in @ Ww + bw) @ Wo + bo ----
  {
    const int MQ = L_ * B_;                       // 24576
    const int nq = MQ * (H_ + LAT_);
    k_build_qin<<<blocks(nq), TPB, 0, stream>>>(hs, tree_vec, qin, nq);
    gemm(qin, H_ + LAT_, WwT, 512, MQ, H_ + LAT_, bw, 1, nullptr, qh);
    gemm(qh, 512, WoT, 800, MQ, 512, bo, 0, qlog, nullptr);
    k_qfinal<<<dim3((unsigned)(MQ / 8)), TPB, 0, stream>>>(qlog, wid, accum, MQ);
  }

  // ---- p head: relu(p_in @ Wu + bu) . Us + bs ----
  {
    const int MP = (T_ + 1) * B_;                 // 48128
    const int np = MP * (2 * H_ + LAT_);
    k_build_pin<<<blocks(np), TPB, 0, stream>>>(xbf, hs, tree_vec, pin, np);
    gemm(pin, 2 * H_ + LAT_, WuT, 512, MP, 2 * H_ + LAT_, bu, 1, ph, nullptr);
    k_pfinal<<<dim3((unsigned)(MP / 8)), TPB, 0, stream>>>(ph, Us, bs, accum, MP);
  }

  k_finalize<<<dim3(1), dim3(64), 0, stream>>>(accum, out);
}
